// DetectorHead_38817914421985
// MI455X (gfx1250) — compile-verified, matrix operand order checked by reading, not run on hardware
//
#include <hip/hip_runtime.h>
#include <math.h>

// ---------------- types for WMMA ----------------
typedef __attribute__((ext_vector_type(16))) __bf16 v16bf;
typedef __attribute__((ext_vector_type(8)))  float  v8f;
typedef int v4i_vs __attribute__((vector_size(16)));

#define AS1 __attribute__((address_space(1)))
#define AS3 __attribute__((address_space(3)))

#if __has_builtin(__builtin_amdgcn_global_load_async_to_lds_b128)
#define HAVE_ASYNC 1
#else
#define HAVE_ASYNC 0
#endif

#if __has_builtin(__builtin_amdgcn_s_wait_asynccnt)
#define ASYNC_WAIT() __builtin_amdgcn_s_wait_asynccnt(0)
#else
#define ASYNC_WAIT() asm volatile("s_wait_asynccnt 0x0" ::: "memory")
#endif

// ---------------- problem constants ----------------
#define BB   32
#define CIN  128
#define HC   60
#define WCc  80
#define MOUT 256
#define KA   1152          // 128*9
#define NPIX 4800          // 60*80
#define HH   480
#define WW   640
#define NTOT (BB*HH*WW)    // 9830400
#define KTOP 1024
#define CAP  32768
#define EPSB 1e-5f
#define STAGE_BYTES 20480  // A tile 16KB + B tile 4KB
#define DYN_LDS (2*STAGE_BYTES + 32)

__device__ __forceinline__ unsigned short f2bf(float f) {
  unsigned u = __float_as_uint(f);
  unsigned r = (u + 0x7FFFu + ((u >> 16) & 1u)) >> 16;
  return (unsigned short)r;
}

// ---------------- conversion kernels ----------------
__global__ void k_cvt_bf16(const float* __restrict__ src,
                           unsigned short* __restrict__ dst, int n) {
  int i = blockIdx.x * blockDim.x + threadIdx.x;
  if (i < n) dst[i] = f2bf(src[i]);
}

__global__ void k_prep_wb(const float* __restrict__ w,  // [65][256]
                          unsigned short* __restrict__ dst) {  // [80][256]
  int i = blockIdx.x * blockDim.x + threadIdx.x;
  if (i >= 80 * 256) return;
  int m = i >> 8, k = i & 255;
  dst[i] = (m < 65) ? f2bf(w[m * 256 + k]) : (unsigned short)0;
}

// ---------------- fused convPa+ReLU+BN+convPb+BN+softmax+pixelshuffle ----------------
__global__ void __launch_bounds__(256)
k_fused_head(const unsigned short* __restrict__ xb,    // [B][128][60][80] bf16
             const unsigned short* __restrict__ Wa2,   // [256][1152] bf16
             const unsigned short* __restrict__ Wbb,   // [80][256] bf16
             const float* __restrict__ biasA, const float* __restrict__ gA,
             const float* __restrict__ betA, const float* __restrict__ mA,
             const float* __restrict__ vA,
             const float* __restrict__ biasB, const float* __restrict__ gB,
             const float* __restrict__ betB, const float* __restrict__ mB,
             const float* __restrict__ vB,
             float* __restrict__ Sout) {               // [B][480][640]
  extern __shared__ char dynRaw[];
  char* dyn = (char*)(((size_t)dynRaw + 31) & ~(size_t)31);
  // double-buffered stage: per buffer [256][32] A + [64][32] B
  unsigned short* AsBuf[2] = {(unsigned short*)dyn,
                              (unsigned short*)(dyn + STAGE_BYTES)};
  unsigned short* BsBuf[2] = {(unsigned short*)(dyn + 16384),
                              (unsigned short*)(dyn + STAGE_BYTES + 16384)};
  float* O = (float*)dyn;  // [64][80] f32, reused after phase 1

  __shared__ __align__(32) unsigned short Ps[64 * 256];  // phase-1 out / phase-2 B
  __shared__ float sAl[256], tAl[256], bAl[256];
  __shared__ float sBl[80], tBl[80];

  const int tid = threadIdx.x;
  const int wv  = tid >> 5;
  const int ln  = tid & 31;
  const int hl  = ln >> 4;
  const int lr  = ln & 15;
  const int b   = blockIdx.y;
  const int p0  = blockIdx.x * 64;

  // fold BN params
  if (tid < 256) {
    float s = gA[tid] * rsqrtf(vA[tid] + EPSB);
    sAl[tid] = s;
    tAl[tid] = betA[tid] - mA[tid] * s;
    bAl[tid] = biasA[tid];
  }
  if (tid < 80) {
    if (tid < 65) {
      float s = gB[tid] * rsqrtf(vB[tid] + EPSB);
      sBl[tid] = s;
      tBl[tid] = (biasB[tid] - mB[tid]) * s + betB[tid];
    } else {
      sBl[tid] = 0.f; tBl[tid] = 0.f;
    }
  }

  auto stageA = [&](int k0, unsigned short* As) {
#if HAVE_ASYNC
#pragma unroll
    for (int c = 0; c < 4; ++c)
      __builtin_amdgcn_global_load_async_to_lds_b128(
          (AS1 v4i_vs*)(Wa2 + (size_t)tid * KA + k0 + c * 8),
          (AS3 v4i_vs*)(As + tid * 32 + c * 8), 0, 0);
#else
    const uint4* src = reinterpret_cast<const uint4*>(Wa2 + (size_t)tid * KA + k0);
    uint4* dst = reinterpret_cast<uint4*>(As + tid * 32);
    dst[0] = src[0]; dst[1] = src[1]; dst[2] = src[2]; dst[3] = src[3];
#endif
  };
  auto stageB = [&](int k0, unsigned short* Bs) {
    for (int e = tid; e < 64 * 32; e += 256) {
      int n = e >> 5, kk = e & 31;
      int kg = k0 + kk;
      int ci = kg / 9, tap = kg % 9;
      int g = p0 + n;
      int yc = g / WCc, xc = g % WCc;
      int yy = yc + tap / 3 - 1, xx = xc + tap % 3 - 1;
      unsigned short v = 0;
      if (yy >= 0 && yy < HC && xx >= 0 && xx < WCc)
        v = xb[(((size_t)b * CIN + ci) * HC + yy) * WCc + xx];
      Bs[n * 32 + kk] = v;
    }
  };

  // -------- phase 1: convPa as implicit GEMM, M=256 N=64 K=1152 --------
  v8f acc[2][4];
#pragma unroll
  for (int i = 0; i < 2; ++i)
#pragma unroll
    for (int j = 0; j < 4; ++j) acc[i][j] = (v8f){0,0,0,0,0,0,0,0};

  stageA(0, AsBuf[0]);
  stageB(0, BsBuf[0]);
#if HAVE_ASYNC
  ASYNC_WAIT();
#endif
  __syncthreads();

  const int NSTEP = KA / 32;  // 36
  for (int step = 0; step < NSTEP; ++step) {
    int cur = step & 1, nxt = cur ^ 1;
    if (step + 1 < NSTEP) {
      stageA((step + 1) * 32, AsBuf[nxt]);
      stageB((step + 1) * 32, BsBuf[nxt]);
      if (step + 2 < NSTEP)
        __builtin_prefetch(Wa2 + (size_t)tid * KA + (step + 2) * 32, 0, 3);
    }
    unsigned short* As = AsBuf[cur];
    unsigned short* Bs = BsBuf[cur];
    v16bf afr[2], bfr[4];
#pragma unroll
    for (int mt = 0; mt < 2; ++mt)
      afr[mt] = *reinterpret_cast<const v16bf*>(As + (wv * 32 + mt * 16 + lr) * 32 + hl * 16);
#pragma unroll
    for (int nt = 0; nt < 4; ++nt)
      bfr[nt] = *reinterpret_cast<const v16bf*>(Bs + (nt * 16 + lr) * 32 + hl * 16);
#pragma unroll
    for (int mt = 0; mt < 2; ++mt)
#pragma unroll
      for (int nt = 0; nt < 4; ++nt)
        acc[mt][nt] = __builtin_amdgcn_wmma_f32_16x16x32_bf16(
            false, afr[mt], false, bfr[nt], (short)0, acc[mt][nt], false, false);
#if HAVE_ASYNC
    ASYNC_WAIT();
#endif
    __syncthreads();
  }

  // epilogue 1: bias + relu + BN, write bf16 tile Ps[n][m]
#pragma unroll
  for (int mt = 0; mt < 2; ++mt)
#pragma unroll
    for (int nt = 0; nt < 4; ++nt)
#pragma unroll
      for (int r = 0; r < 8; ++r) {
        int m = wv * 32 + mt * 16 + r + hl * 8;
        int n = nt * 16 + lr;
        float y = acc[mt][nt][r] + bAl[m];
        y = fmaxf(y, 0.f);
        Ps[n * 256 + m] = f2bf(y * sAl[m] + tAl[m]);
      }
  __syncthreads();

  // -------- phase 2: convPb, M=80(65) N=64 K=256 --------
  v8f acc2[3];
  int mt2[3], nt2[3], live[3];
#pragma unroll
  for (int i = 0; i < 3; ++i) {
    int t = wv + 8 * i;
    live[i] = (t < 20);
    mt2[i] = t >> 2;
    nt2[i] = t & 3;
    acc2[i] = (v8f){0,0,0,0,0,0,0,0};
  }
  for (int k0 = 0; k0 < 256; k0 += 32) {
#pragma unroll
    for (int i = 0; i < 3; ++i) {
      if (!live[i]) continue;
      v16bf a2 = *reinterpret_cast<const v16bf*>(Wbb + (mt2[i] * 16 + lr) * 256 + k0 + hl * 16);
      v16bf b2 = *reinterpret_cast<const v16bf*>(Ps + (nt2[i] * 16 + lr) * 256 + k0 + hl * 16);
      acc2[i] = __builtin_amdgcn_wmma_f32_16x16x32_bf16(
          false, a2, false, b2, (short)0, acc2[i], false, false);
    }
  }
#pragma unroll
  for (int i = 0; i < 3; ++i) {
    if (!live[i]) continue;
#pragma unroll
    for (int r = 0; r < 8; ++r) {
      int c = mt2[i] * 16 + r + hl * 8;
      int n = nt2[i] * 16 + lr;
      O[n * 80 + c] = acc2[i][r] * sBl[c] + tBl[c];
    }
  }
  __syncthreads();

  // softmax over 65 channels + pixel shuffle store (drop channel 64)
  if (tid < 64) {
    int n = tid;
    float mx = -INFINITY;
    for (int c = 0; c < 65; ++c) mx = fmaxf(mx, O[n * 80 + c]);
    float s = 0.f;
    for (int c = 0; c < 65; ++c) {
      float e = __expf(O[n * 80 + c] - mx);
      O[n * 80 + c] = e;
      s += e;
    }
    float inv = 1.f / s;
    int g = p0 + n;
    int yc = g / WCc, xc = g % WCc;
    for (int c = 0; c < 64; ++c)
      Sout[((size_t)b * HH + yc * 8 + (c >> 3)) * WW + xc * 8 + (c & 7)] =
          O[n * 80 + c] * inv;
  }
}

// ---------------- 9x9 max pool (stride 1, -inf padding) ----------------
__global__ void __launch_bounds__(256)
k_pool9(const float* __restrict__ in, float* __restrict__ out) {
  __shared__ float t[24][24];
  int b  = blockIdx.z;
  int x0 = blockIdx.x * 16, y0 = blockIdx.y * 16;
  int tid = threadIdx.y * 16 + threadIdx.x;
  for (int e = tid; e < 24 * 24; e += 256) {
    int ly = e / 24, lx = e % 24;
    int gy = y0 + ly - 4, gx = x0 + lx - 4;
    float v = -INFINITY;
    if (gy >= 0 && gy < HH && gx >= 0 && gx < WW)
      v = in[((size_t)b * HH + gy) * WW + gx];
    t[ly][lx] = v;
  }
  __syncthreads();
  int lx = threadIdx.x, ly = threadIdx.y;
  float m = -INFINITY;
#pragma unroll
  for (int dy = 0; dy < 9; ++dy)
#pragma unroll
    for (int dx = 0; dx < 9; ++dx) m = fmaxf(m, t[ly + dy][lx + dx]);
  out[((size_t)b * HH + y0 + ly) * WW + x0 + lx] = m;
}

// ---------------- NMS elementwise kernels ----------------
__global__ void k_mask_eq(const float* S, const float* U, float* M, int n) {
  int i = blockIdx.x * blockDim.x + threadIdx.x;
  if (i < n) M[i] = (S[i] == U[i]) ? 1.f : 0.f;
}
__global__ void k_supp(const float* S, const float* U, float* V,
                       unsigned char* Su, int n) {
  int i = blockIdx.x * blockDim.x + threadIdx.x;
  if (i < n) {
    bool sp = U[i] > 0.f;
    Su[i] = sp ? 1 : 0;
    V[i] = sp ? 0.f : S[i];
  }
}
__global__ void k_update(float* M, const float* V, const float* U,
                         const unsigned char* Su, int n) {
  int i = blockIdx.x * blockDim.x + threadIdx.x;
  if (i < n) {
    bool nm = (V[i] == U[i]) && (Su[i] == 0);
    M[i] = (M[i] > 0.f || nm) ? 1.f : 0.f;
  }
}
__global__ void k_final(const float* S, const float* M, float* V, int n) {
  int i = blockIdx.x * blockDim.x + threadIdx.x;
  if (i < n) V[i] = (M[i] > 0.f) ? S[i] : 0.f;
}

// ---------------- top-k ----------------
__global__ void k_zero_cnt(int* cnt) {
  if (threadIdx.x < BB) cnt[threadIdx.x] = 0;
}

__global__ void k_collect(const float* __restrict__ V,
                          const float* __restrict__ tptr,
                          int* cnt, unsigned long long* cand) {
  int i = blockIdx.x * blockDim.x + threadIdx.x;
  if (i >= NTOT) return;
  int b = i / (HH * WW);
  int r = i % (HH * WW);
  int y = r / WW, x = r % WW;
  float v = V[i];
  float thr = *tptr;
  if (!(y >= 4 && y < HH - 4 && x >= 4 && x < WW - 4 && v > thr)) return;
  unsigned u = __float_as_uint(v);
  unsigned key = (u & 0x80000000u) ? ~u : (u | 0x80000000u);
  int slot = atomicAdd(&cnt[b], 1);
  if (slot < CAP)
    cand[(size_t)b * CAP + slot] = ((unsigned long long)key << 32) | (unsigned)r;
}

__device__ __forceinline__ void sort2048desc(unsigned long long* ld, int t) {
  for (int k = 2; k <= 2048; k <<= 1) {
    for (int j = k >> 1; j > 0; j >>= 1) {
      for (int e = t; e < 2048; e += 1024) {
        int ix = e ^ j;
        if (ix > e) {
          bool asc = ((e & k) == 0);
          asc = !asc;  // overall descending
          unsigned long long a = ld[e], bb = ld[ix];
          if ((a > bb) == asc) { ld[e] = bb; ld[ix] = a; }
        }
      }
      __syncthreads();
    }
  }
}

__global__ void __launch_bounds__(1024)
k_select(const unsigned long long* __restrict__ cand,
         const int* __restrict__ cnt, const float* __restrict__ tptr,
         float* __restrict__ out) {
  __shared__ unsigned long long ld[2048];
  int b = blockIdx.x;
  int t = threadIdx.x;
  int n = cnt[b]; if (n > CAP) n = CAP;
  unsigned un = __float_as_uint(-1.0f);
  unsigned long long NEG1 = ((unsigned long long)(~un) << 32);
  const unsigned long long* cb = cand + (size_t)b * CAP;

  for (int e = t; e < 2048; e += 1024) ld[e] = (e < n) ? cb[e] : NEG1;
  __syncthreads();
  sort2048desc(ld, t);
  for (int base = 2048; base < n; base += 1024) {
    int e = base + t;
    ld[1024 + t] = (e < n) ? cb[e] : NEG1;
    __syncthreads();
    sort2048desc(ld, t);
  }

  unsigned long long kk = ld[t];
  unsigned ko = (unsigned)(kk >> 32);
  unsigned idx = (unsigned)kk;
  unsigned ub = (ko & 0x80000000u) ? (ko & 0x7FFFFFFFu) : ~ko;
  float v = __uint_as_float(ub);
  float thr = *tptr;
  bool valid = v > thr;
  float* kp = out;
  float* sc = out + (size_t)BB * KTOP * 2;
  float* vl = out + (size_t)BB * KTOP * 3;
  size_t o = (size_t)b * KTOP + t;
  kp[o * 2 + 0] = (float)(idx % WW);
  kp[o * 2 + 1] = (float)(idx / WW);
  sc[o] = valid ? v : 0.f;
  vl[o] = valid ? 1.f : 0.f;
}

// ---------------- host launcher ----------------
extern "C" void kernel_launch(void* const* d_in, const int* in_sizes, int n_in,
                              void* d_out, int out_size, void* d_ws, size_t ws_size,
                              hipStream_t stream) {
  const float* x    = (const float*)d_in[0];
  const float* wA   = (const float*)d_in[1];
  const float* bA   = (const float*)d_in[2];
  const float* gA   = (const float*)d_in[3];
  const float* betA = (const float*)d_in[4];
  const float* mA   = (const float*)d_in[5];
  const float* vA   = (const float*)d_in[6];
  const float* wB   = (const float*)d_in[7];
  const float* bB   = (const float*)d_in[8];
  const float* gB   = (const float*)d_in[9];
  const float* betB = (const float*)d_in[10];
  const float* mB   = (const float*)d_in[11];
  const float* vB   = (const float*)d_in[12];
  const float* thr  = (const float*)d_in[14];

  char* p = (char*)d_ws;
  const size_t NX  = (size_t)BB * CIN * HC * WCc;     // 19,660,800
  const size_t NW  = (size_t)MOUT * KA;               // 294,912
  unsigned short* xb  = (unsigned short*)p;                 p += NX * 2;
  unsigned short* Wa2 = (unsigned short*)p;                 p += NW * 2;
  unsigned short* Wbb = (unsigned short*)p;                 p += 80 * 256 * 2;
  p = (char*)(((size_t)p + 255) & ~(size_t)255);
  float* S  = (float*)p;  p += (size_t)NTOT * 4;
  float* M  = (float*)p;  p += (size_t)NTOT * 4;
  float* U  = (float*)p;  p += (size_t)NTOT * 4;
  float* V  = (float*)p;  p += (size_t)NTOT * 4;
  unsigned char* Su = (unsigned char*)p; p += NTOT;
  p = (char*)(((size_t)p + 255) & ~(size_t)255);
  int* cnt = (int*)p;     p += 256;
  unsigned long long* cand = (unsigned long long*)p;

  // 1) precision conversion
  k_cvt_bf16<<<(int)((NX + 255) / 256), 256, 0, stream>>>(x, xb, (int)NX);
  k_cvt_bf16<<<(int)((NW + 255) / 256), 256, 0, stream>>>(wA, Wa2, (int)NW);
  k_prep_wb<<<80, 256, 0, stream>>>(wB, Wbb);

  // 2) fused detector head -> S [B,480,640]
  k_fused_head<<<dim3(NPIX / 64, BB), 256, DYN_LDS, stream>>>(
      xb, Wa2, Wbb, bA, gA, betA, mA, vA, bB, gB, betB, mB, vB, S);

  // 3) simple_nms
  dim3 pg(WW / 16, HH / 16, BB), pb(16, 16);
  int eb = (NTOT + 255) / 256;
  k_pool9<<<pg, pb, 0, stream>>>(S, U);
  k_mask_eq<<<eb, 256, 0, stream>>>(S, U, M, NTOT);
  for (int it = 0; it < 2; ++it) {
    k_pool9<<<pg, pb, 0, stream>>>(M, U);
    k_supp<<<eb, 256, 0, stream>>>(S, U, V, Su, NTOT);
    k_pool9<<<pg, pb, 0, stream>>>(V, U);
    k_update<<<eb, 256, 0, stream>>>(M, V, U, Su, NTOT);
  }
  k_final<<<eb, 256, 0, stream>>>(S, M, V, NTOT);

  // 4) top-k
  k_zero_cnt<<<1, 64, 0, stream>>>(cnt);
  k_collect<<<eb, 256, 0, stream>>>(V, thr, cnt, cand);
  k_select<<<BB, 1024, 0, stream>>>(cand, cnt, thr, (float*)d_out);
}
